// model_controllerMem_25314537242651
// MI455X (gfx1250) — matrix operand order repaired and checked by reading, not compiled, not run on hardware
//
#include <hip/hip_runtime.h>
#include <hip/hip_bf16.h>
#include <math.h>

typedef _Float16 v8h  __attribute__((ext_vector_type(8)));
typedef _Float16 v16h __attribute__((ext_vector_type(16)));
typedef float    v8f  __attribute__((ext_vector_type(8)));

#define WMMA_F16(a, b, c) \
  __builtin_amdgcn_wmma_f32_16x16x32_f16(false, (a), false, (b), (short)0, (c), false, false)

// ---- WMMA fragment loaders (CDNA5 16-bit layouts, wave32) -------------------
// A (16x32 f16): lanes 0-15 -> M=0..15 / K-base 0 ; lanes 16-31 -> K-base 8.
//   VGPR0-3: K = kb..kb+7, VGPR4-7: K = 16+kb..16+kb+7  (kb = (lane>>4)*8)
__device__ __forceinline__ v16h frag_a(const _Float16* base, int stride,
                                       int m0, int k0, int lane) {
  int m  = m0 + (lane & 15);
  int kb = (lane >> 4) * 8;
  const _Float16* p = base + (size_t)m * stride + k0 + kb;
  union { v8h h[2]; v16h v; } u;
  u.h[0] = *(const v8h*)(p);
  u.h[1] = *(const v8h*)(p + 16);
  return u.v;
}

// B (32x16 f16): lanes 0-15 -> N=0..15 / K-base 0 ; lanes 16-31 -> K-base 8 (same K order as A).
// Weight stored row-major [N][K] (out = x @ W^T -> B col n = W row n).
__device__ __forceinline__ v16h frag_b(const _Float16* W, int stride,
                                       int n0, int k0, int lane) {
  int n  = n0 + (lane & 15);
  int kb = (lane >> 4) * 8;
  const _Float16* p = W + (size_t)n * stride + k0 + kb;
  union { v8h h[2]; v16h v; } u;
  u.h[0] = *(const v8h*)(p);
  u.h[1] = *(const v8h*)(p + 16);
  return u.v;
}

__device__ __forceinline__ float sigmf(float x) { return 1.f / (1.f + __expf(-x)); }

// ---- 1) Conv1d(2->128, k=3, pad=1) + ReLU -> f16 emb[(t*1024+b)*128 + d] ----
__global__ void k_conv_embed(const float* __restrict__ past,
                             const float* __restrict__ cw,
                             const float* __restrict__ cb,
                             _Float16* __restrict__ emb) {
  int b = blockIdx.x, d = threadIdx.x;  // 128 threads
  float w[6];
#pragma unroll
  for (int j = 0; j < 6; ++j) w[j] = cw[d * 6 + j];
  float bias = cb[d];
  for (int t = 0; t < 20; ++t) {
    float s = bias;
#pragma unroll
    for (int j = 0; j < 3; ++j) {
      int tt = t + j - 1;
      if (tt >= 0 && tt < 20) {
        s += past[b * 40 + tt * 2 + 0] * w[j];
        s += past[b * 40 + tt * 2 + 1] * w[3 + j];
      }
    }
    emb[(size_t)(t * 1024 + b) * 128 + d] = (_Float16)fmaxf(s, 0.f);
  }
}

// ---- 2) f32 -> f16 repack ---------------------------------------------------
__global__ void k_f32_to_f16(const float* __restrict__ src,
                             _Float16* __restrict__ dst, int n) {
  for (int i = blockIdx.x * blockDim.x + threadIdx.x; i < n;
       i += gridDim.x * blockDim.x)
    dst[i] = (_Float16)src[i];
}

// ---- 3) L2-normalize memory_past rows -> f16 --------------------------------
__global__ void k_norm_rows(const float* __restrict__ src,
                            _Float16* __restrict__ dst) {
  __shared__ float red[128];
  int row = blockIdx.x, tid = threadIdx.x;
  float v = src[(size_t)row * 128 + tid];
  red[tid] = v * v;
  __syncthreads();
  for (int s = 64; s > 0; s >>= 1) {
    if (tid < s) red[tid] += red[tid + s];
    __syncthreads();
  }
  float nrm = fmaxf(sqrtf(red[0]), 1e-12f);
  dst[(size_t)row * 128 + tid] = (_Float16)(v / nrm);
}

// ---- 4) Generic WMMA GEMM: C[MxN](f16) = A[MxK](f16) * B^T (B row-major NxK) -
__global__ void k_gemm_f16(const _Float16* __restrict__ A,
                           const _Float16* __restrict__ B,
                           _Float16* __restrict__ C, int M, int N, int K) {
  int r0 = blockIdx.x * 32;
  int w = threadIdx.x >> 5, lane = threadIdx.x & 31;
  int lh = lane >> 4, ln = lane & 15;
  int N16 = N >> 4, K32 = K >> 5;
  for (int ct = w; ct < N16; ct += 8) {
    for (int rt = 0; rt < 2; ++rt) {
      v8f acc = {};
      for (int kk = 0; kk < K32; ++kk) {
        v16h a = frag_a(A, K, r0 + rt * 16, kk * 32, lane);
        v16h b = frag_b(B, K, ct * 16, kk * 32, lane);
        acc = WMMA_F16(a, b, acc);
      }
#pragma unroll
      for (int i = 0; i < 8; ++i)
        C[(size_t)(r0 + rt * 16 + i + 8 * lh) * N + ct * 16 + ln] =
            (_Float16)acc[i];
    }
  }
}

// ---- 5) Encoder GRU: 64 rows/block persistent over 20 steps -----------------
// gi precomputed (GIenc, f16, no bias). h f16 in LDS. D=128, gates 384.
__global__ void k_encoder(const _Float16* __restrict__ GI,
                          const _Float16* __restrict__ Whh,
                          const float* __restrict__ bih,
                          const float* __restrict__ bhh,
                          float* __restrict__ statePast,
                          _Float16* __restrict__ qn) {
  __shared__ __align__(16) _Float16 h[64 * 128];
  __shared__ float bihL[384], bhhL[384];
  int tid = threadIdx.x, w = tid >> 5, lane = tid & 31;
  int lh = lane >> 4, ln = lane & 15;
  int gr0 = blockIdx.x * 64;
  for (int i = tid; i < 64 * 128; i += 256) h[i] = (_Float16)0.f;
  for (int i = tid; i < 384; i += 256) { bihL[i] = bih[i]; bhhL[i] = bhh[i]; }
  __syncthreads();
  float hn[4][8];
  for (int t = 0; t < 20; ++t) {
#pragma unroll
    for (int uu = 0; uu < 4; ++uu) {
      int u = w + uu * 8, rt = u >> 3, cg = u & 7;
      v8f aR = {}, aZ = {}, aN = {};
      for (int kk = 0; kk < 4; ++kk) {
        v16h a = frag_a(h, 128, rt * 16, kk * 32, lane);
        aR = WMMA_F16(a, frag_b(Whh, 128, cg * 16, kk * 32, lane), aR);
        aZ = WMMA_F16(a, frag_b(Whh, 128, 128 + cg * 16, kk * 32, lane), aZ);
        aN = WMMA_F16(a, frag_b(Whh, 128, 256 + cg * 16, kk * 32, lane), aN);
      }
#pragma unroll
      for (int i = 0; i < 8; ++i) {
        int m = rt * 16 + i + 8 * lh, c = cg * 16 + ln;
        const _Float16* gi = GI + (size_t)(t * 1024 + gr0 + m) * 384;
        float gir = (float)gi[c] + bihL[c];
        float giz = (float)gi[c + 128] + bihL[c + 128];
        float gin = (float)gi[c + 256] + bihL[c + 256];
        float rg = sigmf(gir + aR[i] + bhhL[c]);
        float zg = sigmf(giz + aZ[i] + bhhL[c + 128]);
        float ng = tanhf(gin + rg * (aN[i] + bhhL[c + 256]));
        float ho = (float)h[m * 128 + c];
        hn[uu][i] = (1.f - zg) * ng + zg * ho;
      }
    }
    __syncthreads();
#pragma unroll
    for (int uu = 0; uu < 4; ++uu) {
      int u = w + uu * 8, rt = u >> 3, cg = u & 7;
#pragma unroll
      for (int i = 0; i < 8; ++i)
        h[(rt * 16 + i + 8 * lh) * 128 + cg * 16 + ln] = (_Float16)hn[uu][i];
    }
    __syncthreads();
  }
  if (tid < 64) {  // emit state_past (f32) + l2-normalized query (f16)
    int gr = gr0 + tid;
    float ss = 0.f;
    for (int c = 0; c < 128; ++c) {
      float v = (float)h[tid * 128 + c];
      statePast[(size_t)gr * 128 + c] = v;
      ss += v * v;
    }
    float nrm = fmaxf(sqrtf(ss), 1e-12f);
    for (int c = 0; c < 128; ++c)
      qn[(size_t)gr * 128 + c] = (_Float16)((float)h[tid * 128 + c] / nrm);
  }
}

// ---- 6) cosine scores (WMMA) + distributed top-5 ----------------------------
// 16 threads per query keep partial top-5 over 1/16 slices; merged at the end.
__global__ void k_scores_topk(const _Float16* __restrict__ qn,
                              const _Float16* __restrict__ Mn,
                              int* __restrict__ topk) {
  __shared__ float sc[16 * 512];
  __shared__ float mv[16 * 16 * 5];
  __shared__ int   mi[16 * 16 * 5];
  int tid = threadIdx.x, w = tid >> 5, lane = tid & 31;
  int lh = lane >> 4, ln = lane & 15;
  int q0 = blockIdx.x * 16;
  int q = tid >> 4, s = tid & 15;  // query / slice within query
  v16h af[4];
#pragma unroll
  for (int kk = 0; kk < 4; ++kk) af[kk] = frag_a(qn, 128, q0, kk * 32, lane);
  float bv[5];
  int bi[5];
#pragma unroll
  for (int k = 0; k < 5; ++k) { bv[k] = -1e30f; bi[k] = 0; }
  for (int mc = 0; mc < 32768; mc += 512) {
    for (int ct = w; ct < 32; ct += 8) {
      v8f acc = {};
#pragma unroll
      for (int kk = 0; kk < 4; ++kk)
        acc = WMMA_F16(af[kk], frag_b(Mn, 128, mc + ct * 16, kk * 32, lane), acc);
#pragma unroll
      for (int i = 0; i < 8; ++i)
        sc[(i + 8 * lh) * 512 + ct * 16 + ln] = acc[i];
    }
    __syncthreads();
    const float* row = sc + q * 512 + s * 32;  // all 256 threads scan
    for (int j = 0; j < 32; ++j) {
      float v = row[j];
      if (v > bv[4]) {
        bv[4] = v; bi[4] = mc + s * 32 + j;
#pragma unroll
        for (int p = 4; p > 0; --p)
          if (bv[p] > bv[p - 1]) {
            float tv = bv[p]; bv[p] = bv[p - 1]; bv[p - 1] = tv;
            int ti = bi[p]; bi[p] = bi[p - 1]; bi[p - 1] = ti;
          }
      }
    }
    __syncthreads();
  }
#pragma unroll
  for (int k = 0; k < 5; ++k) { mv[tid * 5 + k] = bv[k]; mi[tid * 5 + k] = bi[k]; }
  __syncthreads();
  if (tid < 16) {  // merge 16 partial lists (80 candidates) per query
    float fv[5]; int fi[5];
#pragma unroll
    for (int k = 0; k < 5; ++k) { fv[k] = -1e30f; fi[k] = 0; }
    for (int s2 = 0; s2 < 16; ++s2)
#pragma unroll
      for (int k = 0; k < 5; ++k) {
        float v = mv[(tid * 16 + s2) * 5 + k];
        int id = mi[(tid * 16 + s2) * 5 + k];
        if (v > fv[4]) {
          fv[4] = v; fi[4] = id;
#pragma unroll
          for (int p = 4; p > 0; --p)
            if (fv[p] > fv[p - 1]) {
              float tv = fv[p]; fv[p] = fv[p - 1]; fv[p - 1] = tv;
              int ti = fi[p]; fi[p] = fi[p - 1]; fi[p - 1] = ti;
            }
        }
      }
#pragma unroll
    for (int k = 0; k < 5; ++k) topk[(q0 + tid) * 5 + k] = fi[k];
  }
}

// ---- 7) info_total = [state_past | memory_fut[idx]] -> f16 [5120][256] ------
__global__ void k_build_info(const float* __restrict__ statePast,
                             const float* __restrict__ memFut,
                             const int* __restrict__ topk,
                             _Float16* __restrict__ info) {
  int bk = blockIdx.x, c = threadIdx.x;  // 256 threads
  int b = bk / 5, k = bk % 5;
  float v;
  if (c < 128) v = statePast[(size_t)b * 128 + c];
  else v = memFut[(size_t)topk[b * 5 + k] * 128 + (c - 128)];
  info[(size_t)bk * 256 + c] = (_Float16)v;
}

// ---- 8) Decoder GRU: 64 rows/block persistent over 40 steps + fused fc ------
// hidden = 256, gates = 768. gi is GI0 at t=0 and just bih afterwards.
__global__ void k_decoder(const _Float16* __restrict__ GI0,
                          const _Float16* __restrict__ Whh,
                          const float* __restrict__ bih,
                          const float* __restrict__ bhh,
                          const float* __restrict__ fcW,
                          const float* __restrict__ fcb,
                          const float* __restrict__ past,
                          float* __restrict__ out) {
  __shared__ __align__(16) _Float16 h[64 * 256];
  __shared__ float pres[64][2];
  __shared__ float bihL[768], bhhL[768];
  __shared__ float fcWL[512];
  __shared__ float fpart[64][4][2];
  int tid = threadIdx.x, w = tid >> 5, lane = tid & 31;
  int lh = lane >> 4, ln = lane & 15;
  int gr0 = blockIdx.x * 64;
  for (int i = tid; i < 64 * 256; i += 256) h[i] = (_Float16)0.f;
  for (int i = tid; i < 768; i += 256) { bihL[i] = bih[i]; bhhL[i] = bhh[i]; }
  for (int i = tid; i < 512; i += 256) fcWL[i] = fcW[i];
  if (tid < 64) {
    int b = (gr0 + tid) / 5;
    pres[tid][0] = past[b * 40 + 38];
    pres[tid][1] = past[b * 40 + 39];
  }
  __syncthreads();
  float f0 = fcb[0], f1 = fcb[1];
  float hn[8][8];
  for (int t = 0; t < 40; ++t) {
#pragma unroll
    for (int uu = 0; uu < 8; ++uu) {
      int u = w + uu * 8, rt = u >> 4, cg = u & 15;
      __builtin_prefetch(Whh + (size_t)(cg * 16) * 256, 0, 3);
      v8f aR = {}, aZ = {}, aN = {};
      for (int kk = 0; kk < 8; ++kk) {
        v16h a = frag_a(h, 256, rt * 16, kk * 32, lane);
        aR = WMMA_F16(a, frag_b(Whh, 256, cg * 16, kk * 32, lane), aR);
        aZ = WMMA_F16(a, frag_b(Whh, 256, 256 + cg * 16, kk * 32, lane), aZ);
        aN = WMMA_F16(a, frag_b(Whh, 256, 512 + cg * 16, kk * 32, lane), aN);
      }
#pragma unroll
      for (int i = 0; i < 8; ++i) {
        int m = rt * 16 + i + 8 * lh, c = cg * 16 + ln;
        int gr = gr0 + m;
        float gir = bihL[c], giz = bihL[c + 256], gin = bihL[c + 512];
        if (t == 0) {
          const _Float16* g = GI0 + (size_t)gr * 768;
          gir += (float)g[c]; giz += (float)g[c + 256]; gin += (float)g[c + 512];
        }
        float rg = sigmf(gir + aR[i] + bhhL[c]);
        float zg = sigmf(giz + aZ[i] + bhhL[c + 256]);
        float ng = tanhf(gin + rg * (aN[i] + bhhL[c + 512]));
        float ho = (float)h[m * 256 + c];
        hn[uu][i] = (1.f - zg) * ng + zg * ho;
      }
    }
    __syncthreads();
#pragma unroll
    for (int uu = 0; uu < 8; ++uu) {
      int u = w + uu * 8, rt = u >> 4, cg = u & 15;
#pragma unroll
      for (int i = 0; i < 8; ++i)
        h[(rt * 16 + i + 8 * lh) * 256 + cg * 16 + ln] = (_Float16)hn[uu][i];
    }
    __syncthreads();
    {  // fused fc: 4 threads per row, partial sums + reduce
      int r = tid >> 2, sg = tid & 3;
      float a0 = 0.f, a1 = 0.f;
      int c0 = sg * 64;
      for (int c = c0; c < c0 + 64; ++c) {
        float hv = (float)h[r * 256 + c];
        a0 += hv * fcWL[c];
        a1 += hv * fcWL[256 + c];
      }
      fpart[r][sg][0] = a0;
      fpart[r][sg][1] = a1;
    }
    __syncthreads();
    if (tid < 64) {
      int gr = gr0 + tid;
      float a0 = fpart[tid][0][0] + fpart[tid][1][0] + fpart[tid][2][0] + fpart[tid][3][0];
      float a1 = fpart[tid][0][1] + fpart[tid][1][1] + fpart[tid][2][1] + fpart[tid][3][1];
      pres[tid][0] += a0 + f0;
      pres[tid][1] += a1 + f1;
      out[(size_t)(gr * 40 + t) * 2 + 0] = pres[tid][0];
      out[(size_t)(gr * 40 + t) * 2 + 1] = pres[tid][1];
    }
    __syncthreads();
  }
}

extern "C" void kernel_launch(void* const* d_in, const int* in_sizes, int n_in,
                              void* d_out, int out_size, void* d_ws,
                              size_t ws_size, hipStream_t stream) {
  const float* past        = (const float*)d_in[0];
  const float* memory_past = (const float*)d_in[1];
  const float* memory_fut  = (const float*)d_in[2];
  const float* conv_w      = (const float*)d_in[3];
  const float* conv_b      = (const float*)d_in[4];
  const float* enc_Wih     = (const float*)d_in[5];
  const float* enc_Whh     = (const float*)d_in[6];
  const float* enc_bih     = (const float*)d_in[7];
  const float* enc_bhh     = (const float*)d_in[8];
  const float* dec_Wih     = (const float*)d_in[9];
  const float* dec_Whh     = (const float*)d_in[10];
  const float* dec_bih     = (const float*)d_in[11];
  const float* dec_bhh     = (const float*)d_in[12];
  const float* fc_W        = (const float*)d_in[13];
  const float* fc_b        = (const float*)d_in[14];
  float* out = (float*)d_out;

  char* ws = (char*)d_ws;
  size_t off = 0;
  auto take = [&](size_t bytes) -> void* {
    void* p = (void*)(ws + off);
    off += (bytes + 255) & ~(size_t)255;
    return p;
  };
  _Float16* embF    = (_Float16*)take((size_t)20480 * 128 * 2);
  _Float16* WihEncF = (_Float16*)take((size_t)384 * 128 * 2);
  _Float16* WhhEncF = (_Float16*)take((size_t)384 * 128 * 2);
  _Float16* WihDecF = (_Float16*)take((size_t)768 * 256 * 2);
  _Float16* WhhDecF = (_Float16*)take((size_t)768 * 256 * 2);
  _Float16* MnF     = (_Float16*)take((size_t)32768 * 128 * 2);
  _Float16* GIencF  = (_Float16*)take((size_t)20480 * 384 * 2);
  float*    stateP  = (float*)take((size_t)1024 * 128 * 4);
  _Float16* qnF     = (_Float16*)take((size_t)1024 * 128 * 2);
  int*      topk    = (int*)take((size_t)1024 * 5 * 4);
  _Float16* infoF   = (_Float16*)take((size_t)5120 * 256 * 2);
  _Float16* GI0F    = (_Float16*)take((size_t)5120 * 768 * 2);

  k_conv_embed<<<1024, 128, 0, stream>>>(past, conv_w, conv_b, embF);
  k_f32_to_f16<<<(384 * 128 + 255) / 256, 256, 0, stream>>>(enc_Wih, WihEncF, 384 * 128);
  k_f32_to_f16<<<(384 * 128 + 255) / 256, 256, 0, stream>>>(enc_Whh, WhhEncF, 384 * 128);
  k_f32_to_f16<<<(768 * 256 + 255) / 256, 256, 0, stream>>>(dec_Wih, WihDecF, 768 * 256);
  k_f32_to_f16<<<(768 * 256 + 255) / 256, 256, 0, stream>>>(dec_Whh, WhhDecF, 768 * 256);
  k_norm_rows<<<32768, 128, 0, stream>>>(memory_past, MnF);

  // gi for all encoder steps: [20*1024, 384] = emb @ enc_Wih^T
  k_gemm_f16<<<20480 / 32, 256, 0, stream>>>(embF, WihEncF, GIencF, 20480, 384, 128);
  k_encoder<<<1024 / 64, 256, 0, stream>>>(GIencF, WhhEncF, enc_bih, enc_bhh, stateP, qnF);
  k_scores_topk<<<1024 / 16, 256, 0, stream>>>(qnF, MnF, topk);
  k_build_info<<<5120, 256, 0, stream>>>(stateP, memory_fut, topk, infoF);
  // decoder t=0 input gates: [5120, 768] = info_total @ dec_Wih^T
  k_gemm_f16<<<5120 / 32, 256, 0, stream>>>(infoF, WihDecF, GI0F, 5120, 768, 256);
  k_decoder<<<5120 / 64, 256, 0, stream>>>(GI0F, WhhDecF, dec_bih, dec_bhh, fc_W, fc_b, past, out);
}
